// SO3Conv_80779744903849
// MI455X (gfx1250) — matrix operand order repaired and checked by reading, not compiled, not run on hardware
//
#include <hip/hip_runtime.h>

typedef __attribute__((ext_vector_type(16))) _Float16 v16h;
typedef __attribute__((ext_vector_type(8)))  float    v8f;
typedef int v4i_gnu __attribute__((vector_size(16)));   // matches builtin param type

#define F_IN   64
#define F_OUT  64
#define NGRID  512
#define IRREP  1771
#define BATCH  256
#define XROW   113344u   // 64*1771, row stride of x/out per batch
#define LDA    40        // padded LDS pitch (halves): conflict-free frag reads
#define LDB    40

__constant__ unsigned c_off[12] = {0, 1, 10, 35, 84, 165, 286, 455, 680, 969, 1330, 1771};

// Async global->LDS (CDNA5) if the toolchain exposes it; else vector copy.
#if defined(__HIP_DEVICE_COMPILE__) && __has_builtin(__builtin_amdgcn_global_load_async_to_lds_b128) && __has_builtin(__builtin_amdgcn_s_wait_asynccnt)
#define HAVE_ASYNC_LDS 1
#else
#define HAVE_ASYNC_LDS 0
#endif

__device__ inline void cp16(_Float16* lds_dst, const _Float16* gsrc) {
#if HAVE_ASYNC_LDS
    __builtin_amdgcn_global_load_async_to_lds_b128(
        (__attribute__((address_space(1))) v4i_gnu*)gsrc,
        (__attribute__((address_space(3))) v4i_gnu*)lds_dst,
        0, 0);
#else
    *(uint4*)lds_dst = *(const uint4*)gsrc;
#endif
}
__device__ inline void cp_fence() {
#if HAVE_ASYNC_LDS
    __builtin_amdgcn_s_wait_asynccnt(0);
#endif
}

// ---- WMMA fragment loaders (ISA 7.12.2, wave32) ---------------------------
__device__ inline v16h load_a_frag(const _Float16* sA, unsigned rowBase, unsigned lane) {
    v16h a;
    const unsigned mloc = lane & 15u;
    const unsigned kb   = (lane >> 4) * 16u;
    const _Float16* ap = &sA[(rowBase + mloc) * LDA];
#pragma unroll
    for (int r = 0; r < 4; ++r) { a[2*r] = ap[kb + 2u*r]; a[2*r+1] = ap[kb + 2u*r + 1u]; }
#pragma unroll
    for (int r = 4; r < 8; ++r) { unsigned k = 8u + kb + 2u*(r-4); a[2*r] = ap[k]; a[2*r+1] = ap[k+1u]; }
    return a;
}
__device__ inline v16h load_b_frag(const _Float16* sB, unsigned colBase, unsigned lane) {
    v16h b;
    const unsigned nloc = lane & 15u;
    const unsigned kb   = (lane >> 4) * 16u;
    const _Float16* bp = &sB[(colBase + nloc) * LDB + kb];
#pragma unroll
    for (int r = 0; r < 8; ++r) { b[2*r] = bp[2*r]; b[2*r+1] = bp[2*r+1]; }
    return b;
}
__device__ inline v8f wmma16(v16h a, v16h b, v8f c) {
    return __builtin_amdgcn_wmma_f32_16x16x32_f16(false, a, false, b, (short)0, c, false, false);
}

// ---------------------------------------------------------------------------
// Kernel 0: pack x into per-degree K-MINOR f16 slabs (layout == LDS layout):
//   xP[16384*coff(l) + R*K2 + K] = f16(x[b, i, off + u*d + m]),
//   R = m*256 + b, K = u*64 + i, K2 = 64*d.
// ---------------------------------------------------------------------------
__global__ __launch_bounds__(256)
void so3_pack_x(const float* __restrict__ X, _Float16* __restrict__ xP) {
    const unsigned l = blockIdx.y, d = 2u*l + 1u;
    if (blockIdx.x >= 64u*d) return;                 // R0 = bx*4 < 256*d
    const unsigned off = c_off[l];
    const unsigned R  = blockIdx.x*4u + (threadIdx.x >> 6);
    const unsigned kl = threadIdx.x & 63u;           // = i
    const unsigned m = R >> 8, b = R & 255u;
    const unsigned wbase = 16384u*off + R*(64u*d);
    const unsigned rbase = b*XROW + off + m + kl*1771u;
    for (unsigned u = 0; u < d; ++u)
        xP[wbase + u*64u + kl] = (_Float16)X[rbase + u*d];
}

// ---------------------------------------------------------------------------
// Kernel 1: psi = W(4096x512) * D(512x1771) / sqrt(512), stored K-minor:
//   psiP[4096*coff(l) + (v*64+o)*K2 + (u*64+i)]  (f16), K2 = 64*d
// Tile 64x64, 128 threads (4 waves as 2x2, each 32x32 -> 4 WMMA/step).
// ---------------------------------------------------------------------------
__global__ __launch_bounds__(128)
void so3_psi_gemm(const float* __restrict__ W, const float* __restrict__ Dm,
                  _Float16* __restrict__ psiP) {
    __shared__ _Float16 sA[64 * LDA];
    __shared__ _Float16 sB[64 * LDB];
    const unsigned tid = threadIdx.x, lane = tid & 31u, wv = tid >> 5;
    const unsigned wr = wv >> 1, wc = wv & 1u;
    const unsigned tileN = blockIdx.x * 64u, tileM = blockIdx.y * 64u;

    v8f a00 = {}, a01 = {}, a10 = {}, a11 = {};

    for (unsigned k0 = 0; k0 < NGRID; k0 += 32u) {
#pragma unroll
        for (int rep = 0; rep < 4; ++rep) {
            unsigned slot = (unsigned)rep * 128u + tid;
            unsigned r = slot >> 3, ch = slot & 7u;
            float4 f = *(const float4*)(W + (tileM + r)*512u + k0 + ch*4u);
            sA[r*LDA + ch*4u + 0u] = (_Float16)f.x;
            sA[r*LDA + ch*4u + 1u] = (_Float16)f.y;
            sA[r*LDA + ch*4u + 2u] = (_Float16)f.z;
            sA[r*LDA + ch*4u + 3u] = (_Float16)f.w;
        }
#pragma unroll
        for (int rep = 0; rep < 4; ++rep) {
            unsigned slot = (unsigned)rep * 128u + tid;
            unsigned kk = slot >> 4, n = (slot & 15u) * 4u;
            unsigned gb = (k0 + kk) * 1771u + tileN + n;
#pragma unroll
            for (int j = 0; j < 4; ++j) {
                float f = (tileN + n + (unsigned)j < IRREP) ? Dm[gb + (unsigned)j] : 0.0f;
                sB[(n + (unsigned)j)*LDB + kk] = (_Float16)f;
            }
        }
        __syncthreads();
        v16h fa0 = load_a_frag(sA, wr*32u,       lane);
        v16h fa1 = load_a_frag(sA, wr*32u + 16u, lane);
        v16h fb0 = load_b_frag(sB, wc*32u,       lane);
        v16h fb1 = load_b_frag(sB, wc*32u + 16u, lane);
        a00 = wmma16(fa0, fb0, a00); a01 = wmma16(fa0, fb1, a01);
        a10 = wmma16(fa1, fb0, a10); a11 = wmma16(fa1, fb1, a11);
        __syncthreads();
    }

    const float scale = 0.04419417382415922f;  // 1/sqrt(512)
    const float* accs[2][2] = {{(const float*)&a00, (const float*)&a01},
                               {(const float*)&a10, (const float*)&a11}};
#pragma unroll
    for (int nb = 0; nb < 2; ++nb) {
        unsigned col = tileN + wc*32u + (unsigned)nb*16u + (lane & 15u);
        if (col >= IRREP) continue;
        unsigned l = 0;                         // degree lookup: 2x per lane
#pragma unroll
        for (int j = 1; j < 11; ++j) l += (col >= c_off[j]) ? 1u : 0u;
        unsigned d = 2u*l + 1u, rel = col - c_off[l];
        unsigned u = rel / d, v = rel - u*d;
        unsigned pb = 4096u*c_off[l] + u*64u;   // + (v*64+o)*K2 + i
        unsigned K2 = 64u*d;
#pragma unroll
        for (int mr = 0; mr < 2; ++mr) {
#pragma unroll
            for (int g = 0; g < 8; ++g) {
                unsigned row = tileM + wr*32u + (unsigned)mr*16u + (unsigned)g + ((lane >> 4)*8u);
                unsigned i = row >> 6, o = row & 63u;
                psiP[pb + (v*64u + o)*K2 + i] = (_Float16)(accs[mr][nb][g] * scale);
            }
        }
    }
}

// ---------------------------------------------------------------------------
// Kernel 2 (hot): per-degree linear f16 GEMM, double-buffered async pipeline:
//   wait(tile t) -> barrier -> issue async(tile t+1 -> other buffer) -> WMMA(t)
// One barrier per K-step; the async engine fills buffer 1-p during compute.
// Tile 128(R) x 64(C), 256 threads (8 waves as 4x2, each 32x32 -> 4 WMMA/step).
// ---------------------------------------------------------------------------
__global__ __launch_bounds__(256)
void so3_block_gemm(const _Float16* __restrict__ xP, const _Float16* __restrict__ psiP,
                    float* __restrict__ out) {
    const unsigned l = blockIdx.z, d = 2u*l + 1u;
    if (blockIdx.x >= d || blockIdx.y >= 2u*d) return;   // whole block: EXEC stays all-ones
    const unsigned off = c_off[l];
    const unsigned K2 = 64u*d;

    __shared__ _Float16 sA[2][128 * LDA];
    __shared__ _Float16 sB[2][64 * LDB];
    const unsigned tid = threadIdx.x, lane = tid & 31u, wv = tid >> 5;
    const unsigned wr = wv >> 1, wc = wv & 1u;            // 4x2 wave grid
    const unsigned tileR = blockIdx.y * 128u, tileC = blockIdx.x * 64u;

    const unsigned a_r = tid >> 2, ch = (tid & 3u) * 8u;  // 16B chunks along K
    const unsigned b_n = tid >> 2;
    const unsigned aOff = 16384u*off + tileR*K2 + a_r*K2 + ch;   // + k0 (+64*K2 for 2nd half)
    const unsigned bOff = 4096u*off  + tileC*K2 + b_n*K2 + ch;   // + k0
    const unsigned la = a_r*LDA + ch, la2 = (a_r + 64u)*LDA + ch, lb = b_n*LDB + ch;

    v8f c00 = {}, c01 = {}, c10 = {}, c11 = {};

    // prologue: tile 0 -> buffer 0
    cp16(&sA[0][la],  xP  + aOff);
    cp16(&sA[0][la2], xP  + aOff + 64u*K2);
    cp16(&sB[0][lb],  psiP + bOff);

    unsigned p = 0;
    for (unsigned k0 = 0; k0 < K2; k0 += 32u) {
        cp_fence();          // this wave's tile-t asyncs have landed
        __syncthreads();     // all waves: tile t in buf p, everyone done reading buf 1-p
        if (k0 + 32u < K2) { // fill the other buffer while we compute
            cp16(&sA[1u - p][la],  xP  + aOff + k0 + 32u);
            cp16(&sA[1u - p][la2], xP  + aOff + 64u*K2 + k0 + 32u);
            cp16(&sB[1u - p][lb],  psiP + bOff + k0 + 32u);
        }
        const _Float16* cA = sA[p];
        const _Float16* cB = sB[p];
        v16h fa0 = load_a_frag(cA, wr*32u,       lane);
        v16h fa1 = load_a_frag(cA, wr*32u + 16u, lane);
        v16h fb0 = load_b_frag(cB, wc*32u,       lane);
        v16h fb1 = load_b_frag(cB, wc*32u + 16u, lane);
        c00 = wmma16(fa0, fb0, c00); c01 = wmma16(fa0, fb1, c01);
        c10 = wmma16(fa1, fb0, c10); c11 = wmma16(fa1, fb1, c11);
        p ^= 1u;
    }

    // Store: v and m are per-block constants -> affine scatter, no div/mod.
    const float alpha = 1.0f / sqrtf(64.0f * (float)d);
    const unsigned b0 = (blockIdx.y & 1u) * 128u;
    const unsigned cbase = off + blockIdx.x * d + (blockIdx.y >> 1);  // off + v*d + m
    const float* accs[2][2] = {{(const float*)&c00, (const float*)&c01},
                               {(const float*)&c10, (const float*)&c11}};
#pragma unroll
    for (int mr = 0; mr < 2; ++mr) {
#pragma unroll
        for (int nb = 0; nb < 2; ++nb) {
#pragma unroll
            for (int g = 0; g < 8; ++g) {
                unsigned rloc = wr*32u + (unsigned)mr*16u + (unsigned)g + ((lane >> 4)*8u);
                unsigned o    = wc*32u + (unsigned)nb*16u + (lane & 15u);
                out[(b0 + rloc) * XROW + o * 1771u + cbase] = alpha * accs[mr][nb][g];
            }
        }
    }
}

extern "C" void kernel_launch(void* const* d_in, const int* in_sizes, int n_in,
                              void* d_out, int out_size, void* d_ws, size_t ws_size,
                              hipStream_t stream) {
    (void)in_sizes; (void)n_in; (void)out_size; (void)ws_size;
    const float* x  = (const float*)d_in[0];   // (256, 64, 1771)
    const float* Dm = (const float*)d_in[1];   // (512, 1771)
    const float* w  = (const float*)d_in[2];   // (64, 64, 512)
    float* out = (float*)d_out;                // (256, 64, 1771)

    _Float16* psiP = (_Float16*)d_ws;                       // 4096*1771 f16 = 14.5 MB
    _Float16* xP   = (_Float16*)d_ws + 4096u * IRREP;       // 16384*1771 f16 = 58 MB

    so3_pack_x<<<dim3(1344, 11, 1), dim3(256, 1, 1), 0, stream>>>(x, xP);
    so3_psi_gemm<<<dim3(28, 64, 1), dim3(128, 1, 1), 0, stream>>>(w, Dm, psiP);
    so3_block_gemm<<<dim3(21, 42, 11), dim3(256, 1, 1), 0, stream>>>(xP, psiP, out);
}
